// CrissCrossAttention_88175678587270
// MI455X (gfx1250) — compile-verified, matrix operand order checked by reading, not run on hardware
//
#include <hip/hip_runtime.h>

// CDNA5 / gfx1250 criss-cross attention, v2.
// All contractions on v_wmma_f32_16x16x32_bf16 (wave32).
// Every WMMA operand's K direction is contiguous in memory/LDS so fragment
// loads are 16B b128 ops (v1 emitted serialized 4B gathers with
// s_wait_loadcnt 0 after every pair). Memory-bound: ~39 GFLOP vs ~0.7 GB
// of HBM traffic @ 23.3 TB/s => ~30us floor; bf16 intermediates halve traffic.

typedef __attribute__((ext_vector_type(16))) __bf16 v16bf;
typedef __attribute__((ext_vector_type(8)))  __bf16 bf8;
typedef __attribute__((ext_vector_type(4)))  __bf16 bf4;
typedef __attribute__((ext_vector_type(8)))  float  v8f;
typedef __attribute__((ext_vector_type(4)))  float  f4;

#define QKV_CH 192
#define CO     64
#define CIN    512
#define HW     16384   // 128*128
#define WDIM   128

// Contiguous-K bf16 fragment: two 16B loads. k0 = (lane>=16)*8 selects the
// K-group split per the 16-bit matrix VGPR layout ({0..7,16..23} vs {8..15,24..31}).
__device__ __forceinline__ v16bf frag_c(const __bf16* base, int k0) {
  bf8 lo = *(const bf8*)(base + k0);
  bf8 hi = *(const bf8*)(base + 16 + k0);
  v16bf f;
#pragma unroll
  for (int j = 0; j < 8; ++j) { f[j] = lo[j]; f[j + 8] = hi[j]; }
  return f;
}

// Contiguous-K fp32 fragment (weights): four b128 loads + cvt to bf16.
__device__ __forceinline__ v16bf frag_f32c(const float* base, int k0) {
  f4 a0 = *(const f4*)(base + k0);
  f4 a1 = *(const f4*)(base + k0 + 4);
  f4 b0 = *(const f4*)(base + 16 + k0);
  f4 b1 = *(const f4*)(base + 16 + k0 + 4);
  v16bf f;
#pragma unroll
  for (int j = 0; j < 4; ++j) {
    f[j]      = (__bf16)a0[j];
    f[j + 4]  = (__bf16)a1[j];
    f[j + 8]  = (__bf16)b0[j];
    f[j + 12] = (__bf16)b1[j];
  }
  return f;
}

// ---------------------------------------------------------------------------
// Kernel 1: qkv = W[192,512] @ x[512, B*HW] + b -> bf16, written in TWO
// layouts: N = [b][c][h][w] (w contiguous) and T = [b][c][w][h] (h contiguous).
// grid (12, 1024), block 256 (8 waves); block = 16 out-ch x 128 pixels
// (one full image row h0). X tile staged via LDS, transposed to [pix][c].
// ---------------------------------------------------------------------------
__global__ __launch_bounds__(256)
void qkv_proj_kernel(const float* __restrict__ x,
                     const float* __restrict__ w,
                     const float* __restrict__ bias,
                     __bf16* __restrict__ qkvN,
                     __bf16* __restrict__ qkvT) {
  __shared__ __bf16 Xs[128][40];   // [pixel][c_local], pitch 80B (16B multiple)

  const int tid  = threadIdx.x;
  const int wave = tid >> 5, lane = tid & 31;
  const int l15  = lane & 15, hf = lane >> 4, k0 = hf * 8;
  const int mbase = blockIdx.x * 16;
  const int p0 = blockIdx.y * 128;          // 128 consecutive pixels = one h row
  const int b  = p0 >> 14, s0 = p0 & 16383;
  const int h0 = s0 >> 7;
  const int cg = tid >> 5, pg = tid & 31;   // staging roles: 4c x 4pix subtile

  v8f acc = {};
  const float* wb = w + (mbase + l15) * CIN;

  for (int kt = 0; kt < CIN / 32; ++kt) {
    // coalesced b128 loads of a 32c x 128pix fp32 tile (4 independent/thread)
    const float* xb = x + (b * CIN + kt * 32 + cg * 4) * HW + s0 + pg * 4;
    f4 v0 = *(const f4*)(xb);
    f4 v1 = *(const f4*)(xb + HW);
    f4 v2 = *(const f4*)(xb + 2 * HW);
    f4 v3 = *(const f4*)(xb + 3 * HW);
    __syncthreads();                        // prior iteration's reads done
#pragma unroll
    for (int pp = 0; pp < 4; ++pp) {
      bf4 o = { (__bf16)v0[pp], (__bf16)v1[pp], (__bf16)v2[pp], (__bf16)v3[pp] };
      *(bf4*)&Xs[pg * 4 + pp][cg * 4] = o;  // transposed: [pix][c]
    }
    __syncthreads();

    v16bf a  = frag_f32c(wb + kt * 32, k0);             // A: weights (global)
    v16bf bm = frag_c(&Xs[wave * 16 + l15][0], k0);     // B: x (LDS, contiguous)
    acc = __builtin_amdgcn_wmma_f32_16x16x32_bf16(false, a, false, bm,
                                                  (short)0, acc, false, false);
  }

  const int wcol = wave * 16 + l15;   // w coordinate of this lane's pixel
#pragma unroll
  for (int r = 0; r < 8; ++r) {
    const int m = mbase + r + 8 * hf;
    const __bf16 val = (__bf16)(acc[r] + bias[m]);
    qkvN[(b * QKV_CH + m) * HW + h0 * WDIM + wcol] = val;  // [h][w]
    qkvT[(b * QKV_CH + m) * HW + wcol * WDIM + h0] = val;  // [w][h]
  }
}

// ---------------------------------------------------------------------------
// Kernels 2/3: per-(b,line) channel-channel attention (64x64 over 128-long line).
// VERT=0: line = column w, reads T layout (contraction axis h contiguous).
// VERT=1: line = row h,    reads N layout (contraction axis w contiguous);
//         also combines: obuf = gamma * (out_h + out_v).
// obuf layout: [b][pixel][c] so each lane's 8 D-values = one 16B store.
// grid 1024 (= b*128 + line), block 256 (8 waves).
// ---------------------------------------------------------------------------
template <int VERT>
__global__ __launch_bounds__(256)
void cc_attn_kernel(const __bf16* __restrict__ Lbuf,
                    __bf16* __restrict__ obuf,
                    const float* __restrict__ gamma_p) {
  __shared__ float  Es[64][65];    // energies fp32
  __shared__ __bf16 Ps[64][72];    // softmax probs, pitch 144B (16B multiple)
  __shared__ __bf16 Vs[128][72];   // V transposed [t][c]

  const int tid  = threadIdx.x;
  const int wave = tid >> 5, lane = tid & 31;
  const int l15  = lane & 15, hf = lane >> 4, k0 = hf * 8;
  const int g = blockIdx.x, b = g >> 7, line = g & 127;
  const __bf16* base = Lbuf + (b * QKV_CH) * HW + line * WDIM;

  // stage V (channels 128..191) transposed into Vs[t][c]; consumed after the
  // softmax barriers, so these ds stores overlap the E-phase WMMAs.
  {
    const int pg = tid & 31, cg = tid >> 5;
    const int p4 = pg * 4, c8 = cg * 8;
    bf4 v[8];
#pragma unroll
    for (int i = 0; i < 8; ++i)
      v[i] = *(const bf4*)(base + (128 + c8 + i) * HW + p4);
#pragma unroll
    for (int pp = 0; pp < 4; ++pp) {
      bf8 o;
#pragma unroll
      for (int i = 0; i < 8; ++i) o[i] = v[i][pp];
      *(bf8*)&Vs[p4 + pp][c8] = o;
    }
  }

  // --- E[64,64] = Q * K^T (contract line axis, len 128, contiguous) ---
#pragma unroll
  for (int t = wave; t < 16; t += 8) {            // 16 tiles, 2 per wave
    const int ti = t >> 2, tj = t & 3;
    v8f acc = {};
    const __bf16* qb = base + (ti * 16 + l15) * HW;        // q: ch 0..63
    const __bf16* kb = base + (64 + tj * 16 + l15) * HW;   // k: ch 64..127
#pragma unroll
    for (int kt = 0; kt < 4; ++kt) {
      v16bf a  = frag_c(qb + kt * 32, k0);
      v16bf bm = frag_c(kb + kt * 32, k0);
      acc = __builtin_amdgcn_wmma_f32_16x16x32_bf16(false, a, false, bm,
                                                    (short)0, acc, false, false);
    }
#pragma unroll
    for (int r = 0; r < 8; ++r)
      Es[ti * 16 + r + 8 * hf][tj * 16 + l15] = acc[r];
  }
  __syncthreads();

  // --- row softmax (fp32) -> P (bf16) ---
  if (tid < 64) {
    float mx = -3.0e38f;
    for (int j = 0; j < 64; ++j) mx = fmaxf(mx, Es[tid][j]);
    float sum = 0.f;
    for (int j = 0; j < 64; ++j) {
      float e = __expf(Es[tid][j] - mx);
      sum += e;
      Es[tid][j] = e;
    }
    const float inv = 1.f / sum;
    for (int j = 0; j < 64; ++j) Ps[tid][j] = (__bf16)(Es[tid][j] * inv);
  }
  __syncthreads();

  const float gm = VERT ? gamma_p[0] : 0.f;

  // --- O[64,128] = P[64,64] * V[64,128], both operands from LDS ---
#pragma unroll
  for (int t = wave * 4; t < wave * 4 + 4; ++t) { // 32 tiles, 4 per wave
    const int qi = t >> 3, ni = t & 7;
    v8f acc = {};
    const __bf16* pb = &Ps[qi * 16 + l15][0];
    const __bf16* vb = &Vs[ni * 16 + l15][0];
#pragma unroll
    for (int kt = 0; kt < 2; ++kt) {
      v16bf a  = frag_c(pb + kt * 32, k0);
      v16bf bm = frag_c(vb + kt * 32, k0);
      acc = __builtin_amdgcn_wmma_f32_16x16x32_bf16(false, a, false, bm,
                                                    (short)0, acc, false, false);
    }
    const int n = ni * 16 + l15;                        // spatial coord on line
    const int s = VERT ? line * WDIM + n : n * WDIM + line;
    __bf16* op = obuf + (size_t)(b * HW + s) * CO + qi * 16 + 8 * hf;
    if (VERT) {
      bf8 old = *(const bf8*)op;
      bf8 o;
#pragma unroll
      for (int r = 0; r < 8; ++r) o[r] = (__bf16)(gm * (acc[r] + (float)old[r]));
      *(bf8*)op = o;
    } else {
      bf8 o;
#pragma unroll
      for (int r = 0; r < 8; ++r) o[r] = (__bf16)acc[r];
      *(bf8*)op = o;
    }
  }
}

// ---------------------------------------------------------------------------
// Kernel 4: out = W[512,64] @ comb[64, B*HW] + b + x -> fp32.
// comb is [pixel][c] so B-fragments are contiguous. grid (32, 1024), block 256.
// ---------------------------------------------------------------------------
__global__ __launch_bounds__(256)
void out_proj_kernel(const __bf16* __restrict__ obuf,
                     const float* __restrict__ w,
                     const float* __restrict__ bias,
                     const float* __restrict__ x,
                     float* __restrict__ out) {
  const int tid  = threadIdx.x;
  const int wave = tid >> 5, lane = tid & 31;
  const int l15  = lane & 15, hf = lane >> 4, k0 = hf * 8;
  const int mbase = blockIdx.x * 16;
  const int p = blockIdx.y * 128 + wave * 16;
  const int b = p >> 14, s = p & 16383;

  v8f acc = {};
  const float*  wb = w + (mbase + l15) * CO;
  const __bf16* ob = obuf + (size_t)(p + l15) * CO;   // pixel-major, c contiguous
#pragma unroll
  for (int kt = 0; kt < 2; ++kt) {
    v16bf a  = frag_f32c(wb + kt * 32, k0);
    v16bf bm = frag_c(ob + kt * 32, k0);
    acc = __builtin_amdgcn_wmma_f32_16x16x32_bf16(false, a, false, bm,
                                                  (short)0, acc, false, false);
  }
#pragma unroll
  for (int r = 0; r < 8; ++r) {
    const int m = mbase + r + 8 * hf;
    const int idx = (b * CIN + m) * HW + s + l15;
    out[idx] = acc[r] + bias[m] + x[idx];
  }
}

// ---------------------------------------------------------------------------
extern "C" void kernel_launch(void* const* d_in, const int* in_sizes, int n_in,
                              void* d_out, int out_size, void* d_ws, size_t ws_size,
                              hipStream_t stream) {
  const float* x     = (const float*)d_in[0];
  const float* qkv_w = (const float*)d_in[1];
  const float* qkv_b = (const float*)d_in[2];
  const float* out_w = (const float*)d_in[3];
  const float* out_b = (const float*)d_in[4];
  const float* gamma = (const float*)d_in[5];
  float*       out   = (float*)d_out;

  // Workspace: qkvN (50.3 MB) + qkvT (50.3 MB) + combined out (16.8 MB)
  __bf16* qkvN = (__bf16*)d_ws;
  __bf16* qkvT = qkvN + (size_t)8 * QKV_CH * HW;
  __bf16* obuf = qkvT + (size_t)8 * QKV_CH * HW;

  qkv_proj_kernel<<<dim3(12, 1024), 256, 0, stream>>>(x, qkv_w, qkv_b, qkvN, qkvT);
  cc_attn_kernel<0><<<dim3(1024), 256, 0, stream>>>(qkvT, obuf, gamma); // horizontal
  cc_attn_kernel<1><<<dim3(1024), 256, 0, stream>>>(qkvN, obuf, gamma); // vertical + combine
  out_proj_kernel<<<dim3(32, 1024), 256, 0, stream>>>(obuf, out_w, out_b, x, out);
}
